// RecommenderRGCN_50663434223999
// MI455X (gfx1250) — matrix operand order repaired and checked by reading, not compiled, not run on hardware
//
#include <hip/hip_runtime.h>
#include <cstdint>
#include <cstddef>

// ---------------------------------------------------------------------------
// RGCN (PyG-style, aggr='mean') on MI455X / gfx1250.
//   per layer: scatter-add x[src] into agg[(dst,rel)], count edges per bin,
//   then out = sum_r (agg_r/cnt_r) @ W_r + x @ Wroot + b  (bf16 WMMA, f32 acc)
//   epilogues: ReLU fused into GEMM; LayerNorm (+residual) as wave32 kernel.
// ---------------------------------------------------------------------------

typedef __attribute__((ext_vector_type(16))) __bf16          v16bf;
typedef __attribute__((ext_vector_type(8)))  float           v8f;

#define NODES 100000
#define NREL  8
#define NEDGE 640000

// native fp32 -> bf16 (RNE): lowers to v_cvt_pk_bf16_f32-class hardware ops
__device__ __forceinline__ __bf16 f2bf(float f) { return (__bf16)f; }

// ---------------------------------------------------------------------------
__global__ void zero_f32(float* __restrict__ p, long long n) {
  long long i = (long long)blockIdx.x * blockDim.x + threadIdx.x;
  if (i < n) p[i] = 0.0f;
}

// one thread per edge: count edges per (dst, rel) bin (identical all layers)
__global__ void edge_count(const int* __restrict__ ei, const int* __restrict__ et,
                           float* __restrict__ cnt) {
  int e = blockIdx.x * blockDim.x + threadIdx.x;
  if (e >= NEDGE) return;
  int dst = ei[NEDGE + e];
  atomicAdd(&cnt[(size_t)dst * NREL + et[e]], 1.0f);
}

// one thread per (edge, 4-float chunk): agg[(dst,rel)] += x[src]
__global__ void edge_scatter(const float* __restrict__ x, const int* __restrict__ ei,
                             const int* __restrict__ et, float* __restrict__ agg,
                             int di) {
  int gid = blockIdx.x * blockDim.x + threadIdx.x;
  int cpe = di >> 2;                      // float4 chunks per edge
  int e = gid / cpe, c = gid - e * cpe;
  if (e >= NEDGE) return;
  int src = ei[e], dst = ei[NEDGE + e], r = et[e];
  const float4 v = *(const float4*)(x + (size_t)src * di + (size_t)c * 4);
  float* p = agg + ((size_t)dst * NREL + r) * di + (size_t)c * 4;
  atomicAdd(p + 0, v.x);
  atomicAdd(p + 1, v.y);
  atomicAdd(p + 2, v.z);
  atomicAdd(p + 3, v.w);
}

// ---------------------------------------------------------------------------
// Pre-pack weights (8 relations + root) as bf16 in WMMA B-fragment layout:
// frag[s][t][j][lane][e] = bf16( Wsrc[k][n] ), k = t*32 + (lane>=16?8:0) + (e<8?e:e+8),
// n = j*16 + (lane&15).  (16-bit B 32x16: lanes 0-15 hold K=0..7,16..23, 16-31 K=8..15,24..31)
__global__ void build_wfrag(const float* __restrict__ W, const float* __restrict__ Wr,
                            __bf16* __restrict__ frag, int di, int dofull) {
  int idx = blockIdx.x * blockDim.x + threadIdx.x;
  int total = 9 * di * dofull;
  if (idx >= total) return;
  int e = idx & 15;
  int rest = idx >> 4;
  int lane = rest & 31; rest >>= 5;
  int NT = dofull >> 4, KT = di >> 5;
  int j = rest % NT; rest /= NT;
  int t = rest % KT;
  int s = rest / KT;
  int k = t * 32 + ((lane >= 16) ? 8 : 0) + ((e < 8) ? e : e + 8);
  int n = j * 16 + (lane & 15);
  float w = (s < 8) ? W[((size_t)s * di + k) * dofull + n]
                    : Wr[(size_t)k * dofull + n];
  frag[idx] = f2bf(w);
}

// ---------------------------------------------------------------------------
// One wave per 16-node block. A fragments (scaled agg means / root x) are
// loaded once per (source, k-tile) and reused across all DO/16 output tiles,
// so the 410MB agg buffer is streamed from HBM exactly once.
template <int DI, int DO, bool RELU>
__global__ __launch_bounds__(32) void rgcn_gemm(
    const float* __restrict__ agg, const float* __restrict__ cnt,
    const float* __restrict__ x, const __bf16* __restrict__ wfrag,
    const float* __restrict__ bias, float* __restrict__ out) {
  constexpr int KT = DI / 32, NT = DO / 16;
  const int lane = threadIdx.x;
  const int m0 = blockIdx.x * 16;
  const int row = m0 + (lane & 15);        // each lane owns exactly one A row
  const int kb = (lane < 16) ? 0 : 8;      // 16-bit A layout half-wave K base

  v8f acc[NT];
  v8f zero = {};
#pragma unroll
  for (int j = 0; j < NT; ++j) acc[j] = zero;

#pragma unroll
  for (int s = 0; s < 9; ++s) {
    const float* base;
    float scale;
    if (s < 8) {
      float c = cnt[(size_t)row * NREL + s];
      scale = 1.0f / fmaxf(c, 1.0f);                 // mean = sum / max(cnt,1)
      base = agg + ((size_t)row * NREL + s) * DI;
    } else {
      scale = 1.0f;                                   // root term: x @ Wroot
      base = x + (size_t)row * DI;
    }
#pragma unroll
    for (int t = 0; t < KT; ++t) {
      // A 16x32 bf16 layout: lane<16: e0..7 -> K=kb+0..7, e8..15 -> K=kb+16..23
      const float* p = base + t * 32 + kb;
      float4 f0 = *(const float4*)(p);
      float4 f1 = *(const float4*)(p + 4);
      float4 f2 = *(const float4*)(p + 16);
      float4 f3 = *(const float4*)(p + 20);
      v16bf a;
      a[0]  = f2bf(f0.x * scale); a[1]  = f2bf(f0.y * scale);
      a[2]  = f2bf(f0.z * scale); a[3]  = f2bf(f0.w * scale);
      a[4]  = f2bf(f1.x * scale); a[5]  = f2bf(f1.y * scale);
      a[6]  = f2bf(f1.z * scale); a[7]  = f2bf(f1.w * scale);
      a[8]  = f2bf(f2.x * scale); a[9]  = f2bf(f2.y * scale);
      a[10] = f2bf(f2.z * scale); a[11] = f2bf(f2.w * scale);
      a[12] = f2bf(f3.x * scale); a[13] = f2bf(f3.y * scale);
      a[14] = f2bf(f3.z * scale); a[15] = f2bf(f3.w * scale);

      const __bf16* wp = wfrag + (((size_t)(s * KT + t) * NT) * 32 + lane) * 16;
#pragma unroll
      for (int j = 0; j < NT; ++j) {
        v16bf b = *(const v16bf*)(wp + (size_t)j * 512);  // 32 lanes * 16 bf16
        acc[j] = __builtin_amdgcn_wmma_f32_16x16x32_bf16(
            false, a, false, b, (short)0, acc[j], false, false);
      }
    }
  }

  // C/D layout: vgpr v -> (M = v + (lane<16?0:8), N = lane&15 within tile)
  const int mrow = m0 + ((lane < 16) ? 0 : 8);
#pragma unroll
  for (int j = 0; j < NT; ++j) {
    const int col = j * 16 + (lane & 15);
    const float bn = bias[col];
#pragma unroll
    for (int v = 0; v < 8; ++v) {
      float r = acc[j][v] + bn;
      if (RELU) r = fmaxf(r, 0.0f);
      out[(size_t)(mrow + v) * DO + col] = r;
    }
  }
}

// ---------------------------------------------------------------------------
// LayerNorm over 128 features; one wave32 per node; optional residual add.
__global__ __launch_bounds__(128) void ln128(const float* __restrict__ in,
                                             const float* __restrict__ res,
                                             const float* __restrict__ g,
                                             const float* __restrict__ b,
                                             float* __restrict__ out) {
  int node = blockIdx.x * 4 + (threadIdx.x >> 5);
  int lane = threadIdx.x & 31;
  float4 v = *(const float4*)(in + (size_t)node * 128 + (size_t)lane * 4);
  if (res) {
    const float4 r = *(const float4*)(res + (size_t)node * 128 + (size_t)lane * 4);
    v.x += r.x; v.y += r.y; v.z += r.z; v.w += r.w;
  }
  float s = v.x + v.y + v.z + v.w;
#pragma unroll
  for (int o = 16; o > 0; o >>= 1) s += __shfl_xor(s, o, 32);
  float mu = s * (1.0f / 128.0f);
  float dx = v.x - mu, dy = v.y - mu, dz = v.z - mu, dw = v.w - mu;
  float q = dx * dx + dy * dy + dz * dz + dw * dw;
#pragma unroll
  for (int o = 16; o > 0; o >>= 1) q += __shfl_xor(q, o, 32);
  float rstd = rsqrtf(q * (1.0f / 128.0f) + 1e-5f);
  const float4 gg = *(const float4*)(g + (size_t)lane * 4);
  const float4 bb = *(const float4*)(b + (size_t)lane * 4);
  float4 o4;
  o4.x = dx * rstd * gg.x + bb.x;
  o4.y = dy * rstd * gg.y + bb.y;
  o4.z = dz * rstd * gg.z + bb.z;
  o4.w = dw * rstd * gg.w + bb.w;
  *(float4*)(out + (size_t)node * 128 + (size_t)lane * 4) = o4;
}

// ---------------------------------------------------------------------------
extern "C" void kernel_launch(void* const* d_in, const int* in_sizes, int n_in,
                              void* d_out, int out_size, void* d_ws, size_t ws_size,
                              hipStream_t stream) {
  (void)in_sizes; (void)n_in; (void)out_size; (void)ws_size;

  const float* emb = (const float*)d_in[0];
  const float* W[4]  = {(const float*)d_in[1], (const float*)d_in[4],
                        (const float*)d_in[7], (const float*)d_in[10]};
  const float* Wr[4] = {(const float*)d_in[2], (const float*)d_in[5],
                        (const float*)d_in[8], (const float*)d_in[11]};
  const float* B[4]  = {(const float*)d_in[3], (const float*)d_in[6],
                        (const float*)d_in[9], (const float*)d_in[12]};
  const float* g1 = (const float*)d_in[13];
  const float* b1 = (const float*)d_in[14];
  const float* g2 = (const float*)d_in[15];
  const float* b2 = (const float*)d_in[16];
  const int* ei = (const int*)d_in[17];
  const int* et = (const int*)d_in[18];
  float* out = (float*)d_out;

  uint8_t* ws = (uint8_t*)d_ws;
  size_t off = 0;
  auto take = [&](size_t bytes) -> void* {
    void* p = ws + off;
    off += (bytes + 255) & ~(size_t)255;
    return p;
  };
  float* agg = (float*)take((size_t)NODES * NREL * 128 * 4);  // 410 MB
  float* cnt = (float*)take((size_t)NODES * NREL * 4);
  float* t0  = (float*)take((size_t)NODES * 128 * 4);
  float* x1  = (float*)take((size_t)NODES * 128 * 4);
  float* x2  = (float*)take((size_t)NODES * 128 * 4);
  float* x3  = (float*)take((size_t)NODES * 64 * 4);
  __bf16* frag = (__bf16*)take((size_t)9 * 128 * 128 * 2);

  dim3 blk256(256);
  auto grid1 = [](long long n, int b) {
    return dim3((unsigned)((n + b - 1) / b));
  };

  // counts are layer-invariant (edge structure fixed)
  zero_f32<<<grid1((long long)NODES * NREL, 256), blk256, 0, stream>>>(
      cnt, (long long)NODES * NREL);
  edge_count<<<grid1(NEDGE, 256), blk256, 0, stream>>>(ei, et, cnt);

  // ---- Layer 1: emb(64) -> 128, ReLU; LN -> x1
  zero_f32<<<grid1((long long)NODES * NREL * 64, 256), blk256, 0, stream>>>(
      agg, (long long)NODES * NREL * 64);
  edge_scatter<<<grid1((long long)NEDGE * 16, 256), blk256, 0, stream>>>(
      emb, ei, et, agg, 64);
  build_wfrag<<<grid1(9 * 64 * 128, 256), blk256, 0, stream>>>(W[0], Wr[0], frag, 64, 128);
  rgcn_gemm<64, 128, true><<<dim3(NODES / 16), dim3(32), 0, stream>>>(
      agg, cnt, emb, frag, B[0], t0);
  ln128<<<dim3(NODES / 4), dim3(128), 0, stream>>>(t0, nullptr, g1, b1, x1);

  // ---- Layer 2: x1(128) -> 128, ReLU; LN(t0 + x1) -> x2
  zero_f32<<<grid1((long long)NODES * NREL * 128, 256), blk256, 0, stream>>>(
      agg, (long long)NODES * NREL * 128);
  edge_scatter<<<grid1((long long)NEDGE * 32, 256), blk256, 0, stream>>>(
      x1, ei, et, agg, 128);
  build_wfrag<<<grid1(9 * 128 * 128, 256), blk256, 0, stream>>>(W[1], Wr[1], frag, 128, 128);
  rgcn_gemm<128, 128, true><<<dim3(NODES / 16), dim3(32), 0, stream>>>(
      agg, cnt, x1, frag, B[1], t0);
  ln128<<<dim3(NODES / 4), dim3(128), 0, stream>>>(t0, x1, g2, b2, x2);

  // ---- Layer 3: x2(128) -> 64, ReLU -> x3
  zero_f32<<<grid1((long long)NODES * NREL * 128, 256), blk256, 0, stream>>>(
      agg, (long long)NODES * NREL * 128);
  edge_scatter<<<grid1((long long)NEDGE * 32, 256), blk256, 0, stream>>>(
      x2, ei, et, agg, 128);
  build_wfrag<<<grid1(9 * 128 * 64, 256), blk256, 0, stream>>>(W[2], Wr[2], frag, 128, 64);
  rgcn_gemm<128, 64, true><<<dim3(NODES / 16), dim3(32), 0, stream>>>(
      agg, cnt, x2, frag, B[2], x3);

  // ---- Layer 4: x3(64) -> 64 (no activation) -> d_out
  zero_f32<<<grid1((long long)NODES * NREL * 64, 256), blk256, 0, stream>>>(
      agg, (long long)NODES * NREL * 64);
  edge_scatter<<<grid1((long long)NEDGE * 16, 256), blk256, 0, stream>>>(
      x3, ei, et, agg, 64);
  build_wfrag<<<grid1(9 * 64 * 64, 256), blk256, 0, stream>>>(W[3], Wr[3], frag, 64, 64);
  rgcn_gemm<64, 64, false><<<dim3(NODES / 16), dim3(32), 0, stream>>>(
      agg, cnt, x3, frag, B[3], out);
}
